// Head_5360119185755
// MI455X (gfx1250) — matrix operand order repaired and checked by reading, not compiled, not run on hardware
//
#include <hip/hip_runtime.h>

// ---------------------------------------------------------------------------
// Single-head causal attention for MI455X (gfx1250, wave32, WMMA).
//   B=4, T=2048, C=1024, D=64
//   Phase 0: W^T (fp32 [C,D] -> bf16 [D,C]) for contiguous B-operand loads
//   Phase 1: fused QKV projection, bf16 WMMA 16x16x32, writes Q,K row-major
//            bf16 and V transposed bf16 (B-operand layout for P@V)
//   Phase 2: flash attention, one wave per 16-row query tile, 32-key chunks
// ---------------------------------------------------------------------------

typedef __attribute__((ext_vector_type(16))) __bf16 v16bf;
typedef __attribute__((ext_vector_type(8)))  __bf16 v8bf;
typedef __attribute__((ext_vector_type(8)))  float  v8f;

static constexpr int BB = 4;
static constexpr int TT = 2048;
static constexpr int CC = 1024;
static constexpr int DD = 64;

static __device__ __forceinline__ v8f wmma_bf16(v16bf a, v16bf b, v8f c) {
  return __builtin_amdgcn_wmma_f32_16x16x32_bf16(false, a, false, b, (short)0,
                                                 c, false, false);
}

static __device__ __forceinline__ float half_reduce_max(float v) {
  v = fmaxf(v, __shfl_xor(v, 1, 32));
  v = fmaxf(v, __shfl_xor(v, 2, 32));
  v = fmaxf(v, __shfl_xor(v, 4, 32));
  v = fmaxf(v, __shfl_xor(v, 8, 32));
  return v;
}
static __device__ __forceinline__ float half_reduce_sum(float v) {
  v += __shfl_xor(v, 1, 32);
  v += __shfl_xor(v, 2, 32);
  v += __shfl_xor(v, 4, 32);
  v += __shfl_xor(v, 8, 32);
  return v;
}

// ---------------------------------------------------------------------------
// Phase 0: transpose + convert weights.  Wt[mat][d][c] = (bf16) W[mat][c][d]
// ---------------------------------------------------------------------------
__global__ __launch_bounds__(256) void wt_kernel(const float* __restrict__ Wq,
                                                 const float* __restrict__ Wk,
                                                 const float* __restrict__ Wv,
                                                 __bf16* __restrict__ Wt) {
  int i = blockIdx.x * 256 + threadIdx.x;       // 0 .. 3*D*C-1 exact
  int mat = i / (DD * CC);
  int rem = i - mat * (DD * CC);
  int d = rem / CC;
  int c = rem - d * CC;
  const float* W = (mat == 0) ? Wq : (mat == 1) ? Wk : Wv;
  Wt[i] = (__bf16)W[c * DD + d];
}

// ---------------------------------------------------------------------------
// Phase 1: QKV projection.  One wave per 16-row tile (512 tiles total).
// ---------------------------------------------------------------------------
__global__ __launch_bounds__(128) void qkv_kernel(
    const float* __restrict__ x, const __bf16* __restrict__ Wt,
    const float* __restrict__ bq, const float* __restrict__ bk,
    const float* __restrict__ bv, __bf16* __restrict__ Qb,
    __bf16* __restrict__ Kb, __bf16* __restrict__ Vt) {
  const int lane = threadIdx.x & 31;
  const int wave = threadIdx.x >> 5;
  const int tile = blockIdx.x * 4 + wave;       // 0..511
  const int l = lane & 15, h = lane >> 4;
  const int batch = tile >> 7;                  // tile / 128
  const int row0 = (tile & 127) << 4;           // query row within batch

  const float* xrow = x + ((size_t)batch * TT + row0 + l) * CC;

  v8f acc[12];
#pragma unroll
  for (int i = 0; i < 12; ++i) acc[i] = v8f{};

  for (int k0 = 0; k0 < CC; k0 += 32) {
    // A operand: x[row0+l][k0+16h .. +16), fp32 -> bf16
    v8f xa = *(const v8f*)(xrow + k0 + 16 * h);
    v8f xb = *(const v8f*)(xrow + k0 + 16 * h + 8);
    v16bf a;
#pragma unroll
    for (int e = 0; e < 8; ++e) {
      a[e] = (__bf16)xa[e];
      a[8 + e] = (__bf16)xb[e];
    }
#pragma unroll
    for (int mat = 0; mat < 3; ++mat) {
      const __bf16* wt = Wt + (size_t)mat * DD * CC;
#pragma unroll
      for (int t = 0; t < 4; ++t) {
        // B operand: col n = 16t+l, k contiguous from W^T row (32B load)
        v16bf b = *(const v16bf*)(wt + (size_t)(16 * t + l) * CC + k0 + 16 * h);
        acc[mat * 4 + t] = wmma_bf16(a, b, acc[mat * 4 + t]);
      }
    }
  }

  // Store: Q,K row-major bf16; V transposed bf16 [batch][d][t].
#pragma unroll
  for (int mat = 0; mat < 3; ++mat) {
    const float* bias = (mat == 0) ? bq : (mat == 1) ? bk : bv;
#pragma unroll
    for (int t = 0; t < 4; ++t) {
      const int col = 16 * t + l;
      const float bb = bias[col];
#pragma unroll
      for (int r = 0; r < 8; ++r) {
        const int row = row0 + 8 * h + r;
        const float v = acc[mat * 4 + t][r] + bb;
        if (mat == 0)
          Qb[((size_t)batch * TT + row) * DD + col] = (__bf16)v;
        else if (mat == 1)
          Kb[((size_t)batch * TT + row) * DD + col] = (__bf16)v;
        else
          Vt[((size_t)batch * DD + col) * TT + row] = (__bf16)v;
      }
    }
  }
}

// ---------------------------------------------------------------------------
// Phase 2: flash attention.  One wave per 16-row query tile, 32-key chunks.
// ---------------------------------------------------------------------------
static constexpr int PROW = 40;  // LDS row stride in bf16 (80 B, 16B aligned)

__global__ __launch_bounds__(128) void attn_kernel(
    const __bf16* __restrict__ Qb, const __bf16* __restrict__ Kb,
    const __bf16* __restrict__ Vt, float* __restrict__ out) {
  __shared__ __align__(16) __bf16 plds[4 * 16 * PROW];

  const int lane = threadIdx.x & 31;
  const int wave = threadIdx.x >> 5;
  const int tile = blockIdx.x * 4 + wave;       // 0..511
  const int l = lane & 15, h = lane >> 4;
  const int batch = tile >> 7;
  const int q0 = (tile & 127) << 4;             // query tile start (in batch)
  __bf16* myp = plds + wave * (16 * PROW);

  const __bf16* Qbase = Qb + (size_t)batch * TT * DD;
  const __bf16* Kbase = Kb + (size_t)batch * TT * DD;
  const __bf16* Vtb = Vt + (size_t)batch * DD * TT;

  // Q tile in A-operand layout: rows = lane%16, two d-halves.
  v16bf qa0, qa1;
  {
    const __bf16* qr = Qbase + (size_t)(q0 + l) * DD;
    v8bf lo0 = *(const v8bf*)(qr + 16 * h);
    v8bf hi0 = *(const v8bf*)(qr + 8 + 16 * h);
    v8bf lo1 = *(const v8bf*)(qr + 32 + 16 * h);
    v8bf hi1 = *(const v8bf*)(qr + 40 + 16 * h);
    qa0 = __builtin_shufflevector(lo0, hi0, 0, 1, 2, 3, 4, 5, 6, 7, 8, 9, 10,
                                  11, 12, 13, 14, 15);
    qa1 = __builtin_shufflevector(lo1, hi1, 0, 1, 2, 3, 4, 5, 6, 7, 8, 9, 10,
                                  11, 12, 13, 14, 15);
  }

  v8f o[4];
#pragma unroll
  for (int t = 0; t < 4; ++t) o[t] = v8f{};
  float m[8], lsum[8];
#pragma unroll
  for (int r = 0; r < 8; ++r) {
    m[r] = -3.0e38f;
    lsum[r] = 0.0f;
  }

  const int nchunks = (q0 + 16 + 31) >> 5;      // keys 0 .. q0+15 inclusive
  for (int j = 0; j < nchunks; ++j) {
    const int jb = j << 5;

    // ---- S = (Q @ K^T) for 32 keys: two 16-key column tiles --------------
    v8f s[2];
#pragma unroll
    for (int nt = 0; nt < 2; ++nt) {
      const __bf16* kr = Kbase + (size_t)(jb + 16 * nt + l) * DD;
      v16bf b0 = *(const v16bf*)(kr + 16 * h);        // d = 0..31
      v16bf b1 = *(const v16bf*)(kr + 32 + 16 * h);   // d = 32..63
      v8f c = v8f{};
      c = wmma_bf16(qa0, b0, c);
      c = wmma_bf16(qa1, b1, c);
      s[nt] = c;
    }

    // ---- scale, causal mask, online softmax ------------------------------
    float alpha[8];
#pragma unroll
    for (int r = 0; r < 8; ++r) {
      const int row = q0 + 8 * h + r;
      const int c0 = jb + l;
      const int c1 = jb + 16 + l;
      float s0 = (c0 <= row) ? s[0][r] * 0.125f : -1.0e30f;
      float s1 = (c1 <= row) ? s[1][r] * 0.125f : -1.0e30f;

      float mx = half_reduce_max(fmaxf(s0, s1));
      float mnew = fmaxf(m[r], mx);
      alpha[r] = __expf(m[r] - mnew);
      m[r] = mnew;

      float p0 = __expf(s0 - mnew);
      float p1 = __expf(s1 - mnew);
      lsum[r] = lsum[r] * alpha[r] + half_reduce_sum(p0 + p1);

      // P to LDS in row-major (C-layout element (row 8h+r, col 16nt+l))
      myp[(8 * h + r) * PROW + l] = (__bf16)p0;
      myp[(8 * h + r) * PROW + 16 + l] = (__bf16)p1;
    }
#pragma unroll
    for (int t = 0; t < 4; ++t)
#pragma unroll
      for (int r = 0; r < 8; ++r) o[t][r] = o[t][r] * alpha[r];

    // DS ops of a wave are in-order; explicit CDNA5 split-counter wait to be
    // safe before cross-lane re-read of P.
    asm volatile("s_wait_dscnt 0" ::: "memory");

    // ---- P back out of LDS in A-operand layout ---------------------------
    v8bf plo = *(const v8bf*)(myp + l * PROW + 16 * h);
    v8bf phi = *(const v8bf*)(myp + l * PROW + 8 + 16 * h);
    v16bf pa = __builtin_shufflevector(plo, phi, 0, 1, 2, 3, 4, 5, 6, 7, 8, 9,
                                       10, 11, 12, 13, 14, 15);

    // ---- O += P @ V : V^T rows are B-operand contiguous ------------------
#pragma unroll
    for (int t = 0; t < 4; ++t) {
      v16bf b = *(const v16bf*)(Vtb + (size_t)(16 * t + l) * TT + jb + 16 * h);
      o[t] = wmma_bf16(pa, b, o[t]);
    }
  }

  // ---- epilogue: out = O / l ---------------------------------------------
#pragma unroll
  for (int t = 0; t < 4; ++t) {
    const int col = 16 * t + l;
#pragma unroll
    for (int r = 0; r < 8; ++r) {
      const int row = q0 + 8 * h + r;
      out[((size_t)batch * TT + row) * DD + col] = o[t][r] / lsum[r];
    }
  }
}

// ---------------------------------------------------------------------------
// Host launcher
// ---------------------------------------------------------------------------
extern "C" void kernel_launch(void* const* d_in, const int* in_sizes, int n_in,
                              void* d_out, int out_size, void* d_ws,
                              size_t ws_size, hipStream_t stream) {
  const float* x = (const float*)d_in[0];
  const float* Wq = (const float*)d_in[1];
  const float* bq = (const float*)d_in[2];
  const float* Wk = (const float*)d_in[3];
  const float* bk = (const float*)d_in[4];
  const float* Wv = (const float*)d_in[5];
  const float* bv = (const float*)d_in[6];

  const size_t QKV_BYTES = (size_t)BB * TT * DD * sizeof(__bf16);  // 1 MiB
  char* ws = (char*)d_ws;
  __bf16* Qb = (__bf16*)(ws);
  __bf16* Kb = (__bf16*)(ws + QKV_BYTES);
  __bf16* Vt = (__bf16*)(ws + 2 * QKV_BYTES);
  __bf16* Wt = (__bf16*)(ws + 3 * QKV_BYTES);   // 3*D*C bf16 = 384 KiB

  // Phase 0: weight transpose+convert (3*D*C = 196608 elements exactly)
  wt_kernel<<<(3 * DD * CC) / 256, 256, 0, stream>>>(Wq, Wk, Wv, Wt);

  // Phase 1: QKV projection (512 row tiles, 4 waves/block)
  qkv_kernel<<<128, 128, 0, stream>>>(x, Wt, bq, bk, bv, Qb, Kb, Vt);

  // Phase 2: flash attention (512 query tiles, 4 waves/block)
  attn_kernel<<<128, 128, 0, stream>>>(Qb, Kb, Vt, (float*)d_out);
}
